// Classifier_56083682951592
// MI455X (gfx1250) — compile-verified, matrix operand order checked by reading, not run on hardware
//
#include <hip/hip_runtime.h>
#include <hip/hip_bf16.h>
#include <stdint.h>

// ---- problem dims (match reference) ----
#define L_     12
#define D_     1024
#define H_     16
#define FF_    4096
#define V_     32000
#define B_     2
#define S_     2048
#define DH_    64
#define M_     (B_ * S_)      // 4096 tokens
#define INTER_ 400
#define NLAB_  5

typedef unsigned short u16;
typedef __attribute__((ext_vector_type(16))) __bf16 v16bf;
typedef __attribute__((ext_vector_type(8)))  __bf16 v8bf;
typedef __attribute__((ext_vector_type(8)))  float  v8f;

union FragBF { v16bf v; v8bf h8[2]; };

// fp32 -> bf16, round-to-nearest-even
static __device__ inline u16 f2bf(float f) {
  union { float f; unsigned u; } c; c.f = f;
  unsigned u = c.u;
  unsigned r = u + 0x7FFFu + ((u >> 16) & 1u);
  return (u16)(r >> 16);
}

static __device__ inline void store_out(float* p, float v) { *p = v; }
static __device__ inline void store_out(u16* p, float v)   { *p = f2bf(v); }

// ---------------------------------------------------------------------------
// Embedding gather: h[t,:] = embed[ids[t],:]  (vectorized float4 copy)
// ---------------------------------------------------------------------------
__global__ __launch_bounds__(256) void embed_kernel(
    const int* __restrict__ ids, const float* __restrict__ emb,
    float* __restrict__ h) {
  const int t = blockIdx.x;
  const size_t src = (size_t)ids[t] * D_;
  const size_t dst = (size_t)t * D_;
  const int i = threadIdx.x * 4;
  *(float4*)(h + dst + i) = *(const float4*)(emb + src + i);
}

// ---------------------------------------------------------------------------
// LayerNorm over D=1024, one block per row, templated output (bf16 or f32)
// ---------------------------------------------------------------------------
template <typename OUT>
__global__ __launch_bounds__(256) void ln_kernel(
    const float* __restrict__ x, const float* __restrict__ g,
    const float* __restrict__ bta, OUT* __restrict__ out) {
  __shared__ float red[256];
  const int tid = threadIdx.x;
  const size_t base = (size_t)blockIdx.x * D_;

  float s = 0.f;
  for (int i = tid; i < D_; i += 256) s += x[base + i];
  red[tid] = s; __syncthreads();
  for (int off = 128; off > 0; off >>= 1) {
    if (tid < off) red[tid] += red[tid + off];
    __syncthreads();
  }
  const float mu = red[0] * (1.0f / D_);
  __syncthreads();

  float sq = 0.f;
  for (int i = tid; i < D_; i += 256) { float d = x[base + i] - mu; sq += d * d; }
  red[tid] = sq; __syncthreads();
  for (int off = 128; off > 0; off >>= 1) {
    if (tid < off) red[tid] += red[tid + off];
    __syncthreads();
  }
  const float rs = rsqrtf(red[0] * (1.0f / D_) + 1e-5f);

  for (int i = tid; i < D_; i += 256) {
    float v = (x[base + i] - mu) * rs * g[i] + bta[i];
    store_out(&out[base + i], v);
  }
}

// ---------------------------------------------------------------------------
// WMMA GEMM: C[M,N] = act(A_bf16[M,K] @ B_f32[K,N] + bias (+ res))
// 128x128 block tile, 8 waves (2x4), each wave 64x32 via 4x2 wmma tiles.
// Double-buffered LDS: one barrier per 32-K step; stage of tile i+1 overlaps
// the 8 WMMAs of tile i. B tile converted fp32->bf16 and stored TRANSPOSED
// ([n][k]) with two K values packed per ds_store_b32, so fragment loads are
// contiguous 16B ds_load_b128 matching the 16-bit WMMA per-lane layout
// (lane half -> K chunks {kh.., 16+kh..}, kh = (lane>>4)*8).
// ACT: 0=none, 1=tanh-gelu.  RES: add res[idx].  OUTBF: bf16 vs f32 output.
// ---------------------------------------------------------------------------
template <int ACT, bool RES, bool OUTBF>
__global__ __launch_bounds__(256) void gemm_bf16(
    const u16* __restrict__ A, const float* __restrict__ Bw,
    const float* __restrict__ bias, const float* __restrict__ res,
    void* __restrict__ outp, int N, int K) {
  __shared__ u16 sA[2][128 * 40];  // [m][k], row stride 40 u16 (16B-aligned)
  __shared__ u16 sB[2][128 * 40];  // [n][k] transposed

  const int tid = threadIdx.x;
  const int lane = tid & 31, wid = tid >> 5;
  const int bm = blockIdx.y * 128, bn = blockIdx.x * 128;
  const int wm = (wid >> 2) * 64, wn = (wid & 3) * 32;
  const int ln15 = lane & 15;
  const int kh = (lane >> 4) * 8;

  v8f acc[4][2] = {};

  const int arow = tid >> 1, aseg = (tid & 1) * 16;  // A stage: 128x32 bf16
  const int bkr2 = (tid & 15) * 2;                   // B stage: k-pair base
  const int bnc  = (tid >> 4) * 8;                   // 8 n-cols per thread

  auto stageA = [&](int k0, int buf) {
    const uint4* ag = (const uint4*)(A + (size_t)(bm + arow) * K + k0 + aseg);
    uint4* as = (uint4*)(sA[buf] + arow * 40 + aseg);
    as[0] = ag[0];
    as[1] = ag[1];
  };
  auto stageB = [&](int k0, int buf) {
    const float* bg0 = Bw + (size_t)(k0 + bkr2) * N + bn + bnc;
    const float* bg1 = bg0 + N;
#pragma unroll
    for (int i = 0; i < 8; ++i) {
      const unsigned pk = (unsigned)f2bf(bg0[i]) | ((unsigned)f2bf(bg1[i]) << 16);
      *(unsigned*)(sB[buf] + (bnc + i) * 40 + bkr2) = pk;  // packed b32 store
    }
  };

  stageA(0, 0);
  stageB(0, 0);
  int buf = 0;

  for (int k0 = 0; k0 < K; k0 += 32) {
    __syncthreads();  // staged tile visible; prior frag reads complete

    FragBF af[4];
#pragma unroll
    for (int mi = 0; mi < 4; ++mi) {
      const u16* p = sA[buf] + (wm + mi * 16 + ln15) * 40;
      af[mi].h8[0] = *(const v8bf*)(p + kh);
      af[mi].h8[1] = *(const v8bf*)(p + 16 + kh);
    }
    FragBF bf2[2];
#pragma unroll
    for (int ni = 0; ni < 2; ++ni) {
      const u16* p = sB[buf] + (wn + ni * 16 + ln15) * 40;
      bf2[ni].h8[0] = *(const v8bf*)(p + kh);
      bf2[ni].h8[1] = *(const v8bf*)(p + 16 + kh);
    }

    if (k0 + 32 < K) {  // overlap next-tile staging with the WMMAs below
      stageA(k0 + 32, buf ^ 1);
      stageB(k0 + 32, buf ^ 1);
      if (k0 + 64 < K)  // CDNA5 global_prefetch_b8 for the tile after that
        __builtin_prefetch(Bw + (size_t)(k0 + 64 + bkr2) * N + bn + bnc, 0, 1);
    }

#pragma unroll
    for (int mi = 0; mi < 4; ++mi)
#pragma unroll
      for (int ni = 0; ni < 2; ++ni)
        acc[mi][ni] = __builtin_amdgcn_wmma_f32_16x16x32_bf16(
            false, af[mi].v, false, bf2[ni].v, (short)0, acc[mi][ni], false, false);
    buf ^= 1;
  }

  // ---- epilogue: C layout: lane n = lane&15, row m = r + 8*(lane>>4) ----
  const int half16 = lane >> 4;
#pragma unroll
  for (int mi = 0; mi < 4; ++mi) {
#pragma unroll
    for (int ni = 0; ni < 2; ++ni) {
      const int col = bn + wn + ni * 16 + ln15;
      const float bval = bias[col];
#pragma unroll
      for (int r = 0; r < 8; ++r) {
        const int row = bm + wm + mi * 16 + r + 8 * half16;
        const size_t idx = (size_t)row * N + col;
        float v = acc[mi][ni][r] + bval;
        if (RES) v += res[idx];
        if (ACT == 1) {
          const float x3 = v * v * v;
          v = 0.5f * v * (1.0f + tanhf(0.7978845608028654f * (v + 0.044715f * x3)));
        }
        if (OUTBF) ((u16*)outp)[idx] = f2bf(v);
        else       ((float*)outp)[idx] = v;
      }
    }
  }
}

// ---------------------------------------------------------------------------
// Flash attention, one wave per (b, h, 16-query tile). Causal + pad mask.
// scores = Q Kt / 8 via wmma (K over DH=64 -> 2 steps); online softmax with
// 16-lane shfl_xor row reductions (rows live in lane-halves exactly as the
// wmma C layout); P re-laid out via LDS into A-fragments; P@V via 4 wmma.
// V^T staged with two keys packed per ds_store_b32 (16 lanes x 2 keys).
// ---------------------------------------------------------------------------
__global__ __launch_bounds__(32) void attn_kernel(
    const u16* __restrict__ qkv,      // [M, 3*D] bf16
    const int* __restrict__ amask,    // [B, S]
    u16* __restrict__ ctx) {          // [M, D]  bf16
  __shared__ u16 sP[16 * 40];   // P tile 16x32, row stride 40 (16B aligned)
  __shared__ u16 sVT[64 * 40];  // V transposed: [dh][key]

  const int lane = threadIdx.x;
  const int qb = blockIdx.x * 16;
  const int h = blockIdx.y;
  const int b = blockIdx.z;
  const int ln15 = lane & 15, half16 = lane >> 4;
  const int kh = half16 * 8;
  const size_t rowstride = 3 * D_;
  const size_t baseTok = (size_t)b * S_;

  // Q fragments (A-matrix, 16x64 -> two K=32 steps)
  FragBF qf[2];
  {
    const u16* qrow = qkv + (baseTok + qb + ln15) * rowstride + h * DH_;
#pragma unroll
    for (int kk = 0; kk < 2; ++kk) {
      qf[kk].h8[0] = *(const v8bf*)(qrow + kk * 32 + kh);
      qf[kk].h8[1] = *(const v8bf*)(qrow + kk * 32 + 16 + kh);
    }
  }

  v8f o[4] = {};
  float mrun[8], lrun[8];
#pragma unroll
  for (int r = 0; r < 8; ++r) { mrun[r] = -1e30f; lrun[r] = 0.0f; }

  const int jend = (qb + 15) >> 5;  // 32-key blocks up to the diagonal
  for (int j = 0; j <= jend; ++j) {
    const int keyBase = j * 32;

    // ---- scores: two 16x16 tiles (keys keyBase..+15, +16..+31) ----
    v8f s[2] = {};
#pragma unroll
    for (int t = 0; t < 2; ++t) {
      const u16* krow =
          qkv + (baseTok + keyBase + t * 16 + ln15) * rowstride + D_ + h * DH_;
#pragma unroll
      for (int kk = 0; kk < 2; ++kk) {
        FragBF kf;
        kf.h8[0] = *(const v8bf*)(krow + kk * 32 + kh);
        kf.h8[1] = *(const v8bf*)(krow + kk * 32 + 16 + kh);
        s[t] = __builtin_amdgcn_wmma_f32_16x16x32_bf16(
            false, qf[kk].v, false, kf.v, (short)0, s[t], false, false);
      }
    }

    // ---- scale + causal/pad mask (C layout: n=lane&15, m=r+8*half) ----
    const int mk0 = amask[baseTok + keyBase + ln15];
    const int mk1 = amask[baseTok + keyBase + 16 + ln15];
#pragma unroll
    for (int t = 0; t < 2; ++t) {
      const int keyg = keyBase + t * 16 + ln15;
      const int mk = t ? mk1 : mk0;
#pragma unroll
      for (int r = 0; r < 8; ++r) {
        const int qg = qb + r + 8 * half16;
        const float sc = s[t][r] * 0.125f;  // 1/sqrt(64)
        s[t][r] = ((keyg <= qg) && (mk != 0)) ? sc : -1e9f;
      }
    }

    // ---- online softmax per row (reduce across the 16-lane half) ----
#pragma unroll
    for (int r = 0; r < 8; ++r) {
      float mx = fmaxf(s[0][r], s[1][r]);
#pragma unroll
      for (int w = 1; w < 16; w <<= 1) mx = fmaxf(mx, __shfl_xor(mx, w, 32));
      const float mn = fmaxf(mrun[r], mx);
      const float p0 = __expf(s[0][r] - mn);
      const float p1 = __expf(s[1][r] - mn);
      s[0][r] = p0; s[1][r] = p1;
      float rsum = p0 + p1;
#pragma unroll
      for (int w = 1; w < 16; w <<= 1) rsum += __shfl_xor(rsum, w, 32);
      const float corr = __expf(mrun[r] - mn);
      lrun[r] = lrun[r] * corr + rsum;
      mrun[r] = mn;
#pragma unroll
      for (int c = 0; c < 4; ++c) o[c][r] *= corr;
    }

    // ---- stage P (bf16) into LDS ----
#pragma unroll
    for (int t = 0; t < 2; ++t)
#pragma unroll
      for (int r = 0; r < 8; ++r)
        sP[(r + 8 * half16) * 40 + t * 16 + ln15] = f2bf(s[t][r]);

    // ---- stage V^T: 16 lanes handle 2 keys each, packed b32 stores ----
    if (lane < 16) {
      const u16* v0 =
          qkv + (baseTok + keyBase + 2 * lane) * rowstride + 2 * D_ + h * DH_;
      const u16* v1 = v0 + rowstride;
#pragma unroll
      for (int dh = 0; dh < 64; ++dh) {
        const unsigned pk = (unsigned)v0[dh] | ((unsigned)v1[dh] << 16);
        *(unsigned*)(sVT + dh * 40 + 2 * lane) = pk;
      }
    }
    __syncthreads();

    // ---- O += P @ V (A = P 16x32, B = V 32x64 in 4 N-chunks) ----
    FragBF pf;
    pf.h8[0] = *(const v8bf*)(sP + ln15 * 40 + kh);
    pf.h8[1] = *(const v8bf*)(sP + ln15 * 40 + 16 + kh);
#pragma unroll
    for (int c = 0; c < 4; ++c) {
      FragBF vf;
      const u16* p = sVT + (c * 16 + ln15) * 40;
      vf.h8[0] = *(const v8bf*)(p + kh);
      vf.h8[1] = *(const v8bf*)(p + 16 + kh);
      o[c] = __builtin_amdgcn_wmma_f32_16x16x32_bf16(
          false, pf.v, false, vf.v, (short)0, o[c], false, false);
    }
    __syncthreads();
  }

  // ---- normalize and write ctx (bf16) ----
#pragma unroll
  for (int c = 0; c < 4; ++c) {
    const int col = h * DH_ + c * 16 + ln15;
#pragma unroll
    for (int r = 0; r < 8; ++r) {
      const int row = qb + r + 8 * half16;
      ctx[(baseTok + row) * D_ + col] = f2bf(o[c][r] / lrun[r]);
    }
  }
}

// ---------------------------------------------------------------------------
// Mean pool over sequence
// ---------------------------------------------------------------------------
__global__ __launch_bounds__(256) void pool_kernel(
    const float* __restrict__ hf, float* __restrict__ pooled) {
  const int b = blockIdx.y;
  const int d = blockIdx.x * 256 + threadIdx.x;
  float s = 0.f;
  for (int t = 0; t < S_; ++t) s += hf[((size_t)b * S_ + t) * D_ + d];
  pooled[b * D_ + d] = s * (1.0f / S_);
}

// ---------------------------------------------------------------------------
// Head: batchnorm(B=2, batch stats) + concat + reducer(leaky) + classifier
// ---------------------------------------------------------------------------
__global__ __launch_bounds__(256) void head_kernel(
    const float* __restrict__ pooled, const float* __restrict__ sent,
    const float* __restrict__ ppl, const float* __restrict__ gma,
    const float* __restrict__ bta, const float* __restrict__ rw,
    const float* __restrict__ rb, const float* __restrict__ cw,
    const float* __restrict__ cb, float* __restrict__ out) {
  __shared__ float sf[2][D_ + 4];
  __shared__ float sh[2][INTER_];
  const int tid = threadIdx.x;

  for (int d = tid; d < D_; d += 256) {
    const float p0 = pooled[d], p1 = pooled[D_ + d];
    const float mu = 0.5f * (p0 + p1);
    const float d0 = p0 - mu, d1 = p1 - mu;
    const float var = 0.5f * (d0 * d0 + d1 * d1);
    const float rs = rsqrtf(var + 1e-5f);
    sf[0][d] = gma[d] * d0 * rs + bta[d];
    sf[1][d] = gma[d] * d1 * rs + bta[d];
  }
  if (tid < 2) {
    sf[tid][D_ + 0] = sent[tid * 3 + 0];
    sf[tid][D_ + 1] = sent[tid * 3 + 1];
    sf[tid][D_ + 2] = sent[tid * 3 + 2];
    sf[tid][D_ + 3] = ppl[tid];
  }
  __syncthreads();

  for (int idx = tid; idx < 2 * INTER_; idx += 256) {
    const int b = idx / INTER_, i = idx % INTER_;
    float a = rb[i];
    for (int jj = 0; jj < D_ + 4; ++jj) a += sf[b][jj] * rw[jj * INTER_ + i];
    sh[b][i] = (a > 0.f) ? a : 0.01f * a;
  }
  __syncthreads();

  if (tid < 2 * NLAB_) {
    const int b = tid / NLAB_, l = tid % NLAB_;
    float a = cb[l];
    for (int i = 0; i < INTER_; ++i) a += sh[b][i] * cw[i * NLAB_ + l];
    out[b * NLAB_ + l] = a;
  }
}

// ---------------------------------------------------------------------------
extern "C" void kernel_launch(void* const* d_in, const int* in_sizes, int n_in,
                              void* d_out, int out_size, void* d_ws,
                              size_t ws_size, hipStream_t stream) {
  (void)in_sizes; (void)n_in; (void)out_size; (void)ws_size;
  const int*   ids       = (const int*)d_in[0];
  const int*   amask     = (const int*)d_in[1];
  const float* sentiment = (const float*)d_in[2];
  const float* perplex   = (const float*)d_in[3];
  const float* embed     = (const float*)d_in[4];
  const float* Wqkv      = (const float*)d_in[5];
  const float* bqkv      = (const float*)d_in[6];
  const float* Wo        = (const float*)d_in[7];
  const float* bo        = (const float*)d_in[8];
  const float* W1        = (const float*)d_in[9];
  const float* b1        = (const float*)d_in[10];
  const float* W2        = (const float*)d_in[11];
  const float* b2        = (const float*)d_in[12];
  const float* ln1g      = (const float*)d_in[13];
  const float* ln1b      = (const float*)d_in[14];
  const float* ln2g      = (const float*)d_in[15];
  const float* ln2b      = (const float*)d_in[16];
  const float* lnfg      = (const float*)d_in[17];
  const float* lnfb      = (const float*)d_in[18];
  const float* bng       = (const float*)d_in[19];
  const float* bnb       = (const float*)d_in[20];
  const float* rw        = (const float*)d_in[21];
  const float* rb        = (const float*)d_in[22];
  const float* cw        = (const float*)d_in[23];
  const float* cb        = (const float*)d_in[24];

  char* ws = (char*)d_ws;
  size_t off = 0;
  auto alloc = [&](size_t bytes) {
    char* p = ws + off;
    off = (off + bytes + 255) & ~(size_t)255;
    return p;
  };
  float* h      = (float*)alloc((size_t)M_ * D_ * 4);
  float* hfin   = (float*)alloc((size_t)M_ * D_ * 4);
  u16*   xbf    = (u16*)  alloc((size_t)M_ * D_ * 2);
  u16*   qkvbf  = (u16*)  alloc((size_t)M_ * 3 * D_ * 2);
  u16*   ctxbf  = (u16*)  alloc((size_t)M_ * D_ * 2);
  u16*   ffbf   = (u16*)  alloc((size_t)M_ * FF_ * 2);
  float* pooled = (float*)alloc((size_t)B_ * D_ * 4);

  embed_kernel<<<M_, 256, 0, stream>>>(ids, embed, h);

  for (int l = 0; l < L_; ++l) {
    // x = LN1(h) -> bf16
    ln_kernel<u16><<<M_, 256, 0, stream>>>(h, ln1g + l * D_, ln1b + l * D_, xbf);
    // qkv = x @ Wqkv + bqkv  -> bf16
    gemm_bf16<0, false, true><<<dim3(3 * D_ / 128, M_ / 128), 256, 0, stream>>>(
        xbf, Wqkv + (size_t)l * D_ * 3 * D_, bqkv + (size_t)l * 3 * D_, nullptr,
        qkvbf, 3 * D_, D_);
    // ctx = attention(q,k,v) -> bf16
    attn_kernel<<<dim3(S_ / 16, H_, B_), 32, 0, stream>>>(qkvbf, amask, ctxbf);
    // h = h + ctx @ Wo + bo
    gemm_bf16<0, true, false><<<dim3(D_ / 128, M_ / 128), 256, 0, stream>>>(
        ctxbf, Wo + (size_t)l * D_ * D_, bo + (size_t)l * D_, h, h, D_, D_);
    // x = LN2(h) -> bf16
    ln_kernel<u16><<<M_, 256, 0, stream>>>(h, ln2g + l * D_, ln2b + l * D_, xbf);
    // ff = gelu(x @ W1 + b1) -> bf16
    gemm_bf16<1, false, true><<<dim3(FF_ / 128, M_ / 128), 256, 0, stream>>>(
        xbf, W1 + (size_t)l * D_ * FF_, b1 + (size_t)l * FF_, nullptr, ffbf,
        FF_, D_);
    // h = h + ff @ W2 + b2
    gemm_bf16<0, true, false><<<dim3(D_ / 128, M_ / 128), 256, 0, stream>>>(
        ffbf, W2 + (size_t)l * FF_ * D_, b2 + (size_t)l * D_, h, h, D_, FF_);
  }

  ln_kernel<float><<<M_, 256, 0, stream>>>(h, lnfg, lnfb, hfin);
  pool_kernel<<<dim3(D_ / 256, B_), 256, 0, stream>>>(hfin, pooled);
  head_kernel<<<1, 256, 0, stream>>>(pooled, sentiment, perplex, bng, bnb, rw,
                                     rb, cw, cb, (float*)d_out);
}
